// KSR_3676492005472
// MI455X (gfx1250) — matrix operand order repaired and checked by reading, not compiled, not run on hardware
//
#include <hip/hip_runtime.h>
#include <hip/hip_bf16.h>
#include <math.h>

typedef _Float16 v16h __attribute__((ext_vector_type(16)));
typedef _Float16 h8   __attribute__((ext_vector_type(8)));
typedef float    v8f  __attribute__((ext_vector_type(8)));

static constexpr int BATCH   = 1024;
static constexpr int SEQL    = 50;
static constexpr int NITEMS  = 100000;
static constexpr int EH      = 64;     // E == H == D == 64
static constexpr int RN      = 16;     // relations used (17 - 1)
static constexpr float GAMMA = 0.5f;

__device__ __forceinline__ float sigmoidf_(float x) { return 1.f / (1.f + __expf(-x)); }

__device__ __forceinline__ v8f wmma16(v16h a, v16h b, v8f c) {
    return __builtin_amdgcn_wmma_f32_16x16x32_f16(false, a, false, b, (short)0, c, false, false);
}

// A fragment: 16x32 f16 tile, row = lane&15, chunks at K = off + 8*(lane>>4) and +16.
__device__ __forceinline__ v16h load_a16(const _Float16* base, int rowStride, int lane, int kOff) {
    const _Float16* p = base + (lane & 15) * rowStride + kOff + 8 * (lane >> 4);
    union { v16h v; h8 h[2]; } u;
    u.h[0] = *reinterpret_cast<const h8*>(p);
    u.h[1] = *reinterpret_cast<const h8*>(p + 16);
    return u.v;
}

// B fragment: 32x16 tile sourced from row-major W (rows are the N dim):
// lane holds row n = lane&15, chunks at K = off + 8*(lane>>4) and +16.
__device__ __forceinline__ v16h load_b16(const _Float16* base, int rowStride, int lane, int kOff) {
    const _Float16* p = base + (lane & 15) * rowStride + kOff + 8 * (lane >> 4);
    union { v16h v; h8 h[2]; } u;
    u.h[0] = *reinterpret_cast<const h8*>(p);
    u.h[1] = *reinterpret_cast<const h8*>(p + 16);
    return u.v;
}

// ---------------------------------------------------------------------------
// Kernel 1: fused embedding-gather + GRU scan; one wave handles 16 batch rows.
// ---------------------------------------------------------------------------
__global__ __launch_bounds__(32) void gru_kernel(const int* __restrict__ seq,
                                                 const float* __restrict__ item_emb,
                                                 const float* __restrict__ W_ih,
                                                 const float* __restrict__ W_hh,
                                                 float* __restrict__ seqH) {
    __shared__ _Float16 sWih[192 * 64];
    __shared__ _Float16 sWhh[192 * 64];
    __shared__ _Float16 sEmb[16 * 64];
    __shared__ _Float16 sH[16 * 64];
    __shared__ int sLi[16];

    const int lane = threadIdx.x;
    const int r0   = blockIdx.x * 16;

    for (int i = lane; i < 192 * 64; i += 32) {
        sWih[i] = (_Float16)W_ih[i];
        sWhh[i] = (_Float16)W_hh[i];
    }
    for (int i = lane; i < 16 * 64; i += 32) sH[i] = (_Float16)0.f;
    if (lane < 16) {
        int cnt = 0;
        for (int t = 0; t < SEQL; ++t) cnt += (seq[(r0 + lane) * SEQL + t] != 0) ? 1 : 0;
        sLi[lane] = (cnt - 1 > 0) ? (cnt - 1) : 0;
    }
    __syncthreads();

    const int hi = lane >> 4, lo = lane & 15;

    for (int t = 0; t < SEQL; ++t) {
        // gather masked item embeddings for this step (f32 -> f16 tile in LDS)
        for (int i = lane; i < 16 * 64; i += 32) {
            int row = i >> 6, col = i & 63;
            int it  = seq[(r0 + row) * SEQL + t];
            float v = 0.f;
            if (it > 0) v = item_emb[(it - 1) * 64 + col];
            sEmb[i] = (_Float16)v;
        }
        __syncthreads();

        v8f accR[4], accZ[4], accN[4], accHN[4];
        #pragma unroll
        for (int kt = 0; kt < 4; ++kt) { accR[kt] = (v8f)(0.f); accZ[kt] = (v8f)(0.f); accN[kt] = (v8f)(0.f); accHN[kt] = (v8f)(0.f); }

        #pragma unroll
        for (int kt = 0; kt < 4; ++kt) {
            #pragma unroll
            for (int ko = 0; ko < 2; ++ko) {
                v16h aE = load_a16(sEmb, 64, lane, ko * 32);
                v16h aH = load_a16(sH,   64, lane, ko * 32);
                // r gate: rows [0,64)
                accR[kt]  = wmma16(aE, load_b16(&sWih[(kt * 16) * 64],        64, lane, ko * 32), accR[kt]);
                accR[kt]  = wmma16(aH, load_b16(&sWhh[(kt * 16) * 64],        64, lane, ko * 32), accR[kt]);
                // z gate: rows [64,128)
                accZ[kt]  = wmma16(aE, load_b16(&sWih[(64 + kt * 16) * 64],   64, lane, ko * 32), accZ[kt]);
                accZ[kt]  = wmma16(aH, load_b16(&sWhh[(64 + kt * 16) * 64],   64, lane, ko * 32), accZ[kt]);
                // n gate: rows [128,192); keep gi_n and gh_n separate
                accN[kt]  = wmma16(aE, load_b16(&sWih[(128 + kt * 16) * 64],  64, lane, ko * 32), accN[kt]);
                accHN[kt] = wmma16(aH, load_b16(&sWhh[(128 + kt * 16) * 64],  64, lane, ko * 32), accHN[kt]);
            }
        }

        // elementwise gate math + hidden update (C/D layout: m = g + 8*hi, n = lo)
        #pragma unroll
        for (int kt = 0; kt < 4; ++kt) {
            #pragma unroll
            for (int g = 0; g < 8; ++g) {
                int m = g + 8 * hi;
                int k = kt * 16 + lo;
                float r  = sigmoidf_(accR[kt][g]);
                float z  = sigmoidf_(accZ[kt][g]);
                float nn = tanhf(accN[kt][g] + r * accHN[kt][g]);
                float ho = (float)sH[m * 64 + k];
                float hn = (1.f - z) * nn + z * ho;
                sH[m * 64 + k] = (_Float16)hn;
                if (t == sLi[m]) seqH[(r0 + m) * 64 + k] = hn;
            }
        }
        __syncthreads();
    }
}

// ---------------------------------------------------------------------------
// Kernel 2: KG memory scan; one wave per batch row, 2 lanes per relation.
// ---------------------------------------------------------------------------
__global__ __launch_bounds__(256) void memscan_kernel(const int* __restrict__ seq,
                                                      const int* __restrict__ i2e,
                                                      const float* __restrict__ ent_emb,
                                                      const float* __restrict__ rel_emb,
                                                      float* __restrict__ umem) {
    __shared__ float sHead[8][64];
    const int lane = threadIdx.x & 31;
    const int wave = threadIdx.x >> 5;
    const int b    = blockIdx.x * 8 + wave;
    const int r    = lane >> 1;
    const int dbase = (lane & 1) * 32;

    float mem[32], relv[32];
    #pragma unroll
    for (int j = 0; j < 32; ++j) {
        mem[j]  = 0.f;
        relv[j] = rel_emb[(1 + r) * 64 + dbase + j];
    }
    int cnt = 0;
    for (int t = 0; t < SEQL; ++t) cnt += (seq[b * SEQL + t] != 0) ? 1 : 0;
    const int li = (cnt - 1 > 0) ? (cnt - 1) : 0;

    for (int t = 0; t < SEQL; ++t) {
        int it = seq[b * SEQL + t];
        {
            int d0 = lane * 2;
            float h0 = 0.f, h1 = 0.f;
            if (it > 0) {
                int ent = i2e[it - 1];
                if (ent >= 0) { h0 = ent_emb[ent * 64 + d0]; h1 = ent_emb[ent * 64 + d0 + 1]; }
            }
            sHead[wave][d0] = h0;
            sHead[wave][d0 + 1] = h1;
        }
        __syncthreads();
        float tail[32];
        float dot = 0.f;
        #pragma unroll
        for (int j = 0; j < 32; ++j) {
            tail[j] = sHead[wave][dbase + j] + relv[j];
            dot += mem[j] * tail[j];
        }
        dot += __shfl_xor(dot, 1, 32);
        float z = sigmoidf_(dot);
        #pragma unroll
        for (int j = 0; j < 32; ++j) mem[j] = (1.f - z) * mem[j] + z * tail[j];
        if (t == li) {
            #pragma unroll
            for (int j = 0; j < 32; ++j) umem[b * (RN * 64) + r * 64 + dbase + j] = mem[j];
        }
        __syncthreads();
    }
}

// ---------------------------------------------------------------------------
// Kernel 3: q_k, softmax attention over relations, u_m, p_u (emitted as f16).
// ---------------------------------------------------------------------------
__global__ __launch_bounds__(64) void attn_kernel(const float* __restrict__ seqH,
                                                  const float* __restrict__ rel_emb,
                                                  const float* __restrict__ umem,
                                                  const float* __restrict__ Wk,
                                                  const float* __restrict__ bk,
                                                  const float* __restrict__ Wu,
                                                  const float* __restrict__ bu,
                                                  _Float16* __restrict__ pu16) {
    __shared__ float sh[64], sqk[64], slog[16], sattn[16], sum_[64];
    const int b = blockIdx.x;
    const int j = threadIdx.x;

    sh[j] = seqH[b * 64 + j];
    __syncthreads();

    float q = bk[j];
    for (int k = 0; k < 64; ++k) q += sh[k] * Wk[j * 64 + k];
    sqk[j] = q;
    __syncthreads();

    if (j < 16) {
        float d = 0.f;
        for (int k = 0; k < 64; ++k) d += sqk[k] * rel_emb[(1 + j) * 64 + k];
        slog[j] = GAMMA * d;
    }
    __syncthreads();
    if (j < 16) {
        float mx = -1e30f;
        for (int rr = 0; rr < 16; ++rr) mx = fmaxf(mx, slog[rr]);
        float s = 0.f;
        for (int rr = 0; rr < 16; ++rr) s += __expf(slog[rr] - mx);
        sattn[j] = __expf(slog[j] - mx) / s;
    }
    __syncthreads();

    {
        float u = 0.f;
        for (int rr = 0; rr < 16; ++rr) u += sattn[rr] * umem[b * (RN * 64) + rr * 64 + j];
        sum_[j] = u;
    }
    __syncthreads();

    float p = bu[j];
    for (int k = 0; k < 64; ++k) p += sh[k]   * Wu[j * 128 + k];
    for (int k = 0; k < 64; ++k) p += sum_[k] * Wu[j * 128 + 64 + k];
    pu16[b * 64 + j] = (_Float16)p;
}

// ---------------------------------------------------------------------------
// Kernel 4: all_q_i = [item_emb | h_e] @ Wi^T + bi, stored as f16 (100000x64).
// Block = 128 threads (4 waves); 64 item rows per block; WMMA over K=128.
// ---------------------------------------------------------------------------
__global__ __launch_bounds__(128) void qi_kernel(const float* __restrict__ item_emb,
                                                 const int* __restrict__ i2e,
                                                 const float* __restrict__ ent_emb,
                                                 const float* __restrict__ Wi,
                                                 const float* __restrict__ bi,
                                                 _Float16* __restrict__ q16) {
    __shared__ _Float16 sA[64 * 128];
    __shared__ _Float16 sW[64 * 128];
    __shared__ float sBi[64];
    const int tid = threadIdx.x, lane = tid & 31, wave = tid >> 5;
    const int g0 = blockIdx.x * 64;

    for (int i = tid; i < 64 * 128; i += 128) sW[i] = (_Float16)Wi[i];
    if (tid < 64) sBi[tid] = bi[tid];
    for (int i = tid; i < 64 * 128; i += 128) {
        int row = i >> 7, col = i & 127;
        int g = g0 + row;
        float v = 0.f;
        if (g < NITEMS) {
            if (col < 64) v = item_emb[g * 64 + col];
            else {
                int ent = i2e[g];
                if (ent >= 0) v = ent_emb[ent * 64 + (col - 64)];
            }
        }
        sA[i] = (_Float16)v;
    }
    __syncthreads();

    const int hi = lane >> 4, lo = lane & 15;
    const _Float16* aBase = &sA[(wave * 16) * 128];
    #pragma unroll
    for (int nt = 0; nt < 4; ++nt) {
        v8f acc = (v8f)(0.f);
        #pragma unroll
        for (int ko = 0; ko < 4; ++ko) {
            v16h a  = load_a16(aBase, 128, lane, ko * 32);
            v16h bb = load_b16(&sW[(nt * 16) * 128], 128, lane, ko * 32);
            acc = wmma16(a, bb, acc);
        }
        #pragma unroll
        for (int g8 = 0; g8 < 8; ++g8) {
            int m = g8 + 8 * hi;
            int grow = g0 + wave * 16 + m;
            int col = nt * 16 + lo;
            if (grow < NITEMS) q16[grow * 64 + col] = (_Float16)(acc[g8] + sBi[col]);
        }
    }
}

// ---------------------------------------------------------------------------
// Kernel 5: scores = p_u @ all_q_i^T -> (1024 x 100000) f32. Store-BW bound.
// grid = (782, 16); block = 256 (8 waves). Each wave: one n-tile, FOUR m-tiles
// per B-fragment load (4x L2 read-traffic reduction). B tile prefetched before
// the A-staging barrier (global_prefetch_b8).
// ---------------------------------------------------------------------------
__global__ __launch_bounds__(256) void score_kernel(const _Float16* __restrict__ pu16,
                                                    const _Float16* __restrict__ q16,
                                                    float* __restrict__ out) {
    __shared__ _Float16 sA[64 * 64];   // 64 rows of p_u (f16), 8 KB
    const int tid = threadIdx.x, lane = tid & 31, wave = tid >> 5;
    const int m0    = blockIdx.y * 64;
    const int ntile = blockIdx.x * 8 + wave;
    const bool nvalid = (ntile < NITEMS / 16);   // wave-uniform
    const _Float16* bBase = q16 + (ntile * 16) * 64;  // B cols = q_i rows (row-major)

    if (nvalid) {
        // 32 lanes x 64B covers the whole 2 KB B tile; overlaps with LDS fill.
        const _Float16* pf = bBase + (lane & 15) * 64 + (lane >> 4) * 32;
        __builtin_prefetch(pf, 0, 0);
    }
    for (int i = tid; i < 64 * 64; i += 256) sA[i] = pu16[m0 * 64 + i];
    __syncthreads();

    if (nvalid) {   // EXEC stays all-1 inside (uniform branch)
        const int hi = lane >> 4, lo = lane & 15;
        v8f acc[4];
        #pragma unroll
        for (int mq = 0; mq < 4; ++mq) acc[mq] = (v8f)(0.f);
        #pragma unroll
        for (int ko = 0; ko < 2; ++ko) {
            v16h bb = load_b16(bBase, 64, lane, ko * 32);
            #pragma unroll
            for (int mq = 0; mq < 4; ++mq) {
                v16h a = load_a16(&sA[(mq * 16) * 64], 64, lane, ko * 32);
                acc[mq] = wmma16(a, bb, acc[mq]);
            }
        }
        #pragma unroll
        for (int mq = 0; mq < 4; ++mq) {
            #pragma unroll
            for (int g8 = 0; g8 < 8; ++g8) {
                int m = m0 + mq * 16 + g8 + 8 * hi;
                int n = ntile * 16 + lo;
                out[(size_t)m * NITEMS + n] = acc[mq][g8];
            }
        }
    }
}

// ---------------------------------------------------------------------------

extern "C" void kernel_launch(void* const* d_in, const int* in_sizes, int n_in,
                              void* d_out, int out_size, void* d_ws, size_t ws_size,
                              hipStream_t stream) {
    const int*   seq      = (const int*)d_in[0];
    const int*   i2e      = (const int*)d_in[1];
    const float* item_emb = (const float*)d_in[2];
    const float* ent_emb  = (const float*)d_in[3];
    const float* rel_emb  = (const float*)d_in[4];
    const float* W_ih     = (const float*)d_in[5];
    const float* W_hh     = (const float*)d_in[6];
    const float* Wk       = (const float*)d_in[7];
    const float* bk       = (const float*)d_in[8];
    const float* Wu       = (const float*)d_in[9];
    const float* bu       = (const float*)d_in[10];
    const float* Wi       = (const float*)d_in[11];
    const float* bi       = (const float*)d_in[12];
    float* out = (float*)d_out;

    char* ws = (char*)d_ws;
    float*    seqH = (float*)(ws);                                // 1024*64*4      = 256 KB
    float*    umem = (float*)(ws + 262144);                       // 1024*16*64*4   = 4 MB
    _Float16* pu16 = (_Float16*)(ws + 262144 + 4194304);          // 1024*64*2      = 128 KB
    _Float16* q16  = (_Float16*)(ws + 262144 + 4194304 + 131072); // 100000*64*2    = 12.8 MB

    gru_kernel<<<BATCH / 16, 32, 0, stream>>>(seq, item_emb, W_ih, W_hh, seqH);
    memscan_kernel<<<BATCH / 8, 256, 0, stream>>>(seq, i2e, ent_emb, rel_emb, umem);
    attn_kernel<<<BATCH, 64, 0, stream>>>(seqH, rel_emb, umem, Wk, bk, Wu, bu, pu16);
    qi_kernel<<<(NITEMS + 63) / 64, 128, 0, stream>>>(item_emb, i2e, ent_emb, Wi, bi, q16);
    score_kernel<<<dim3((NITEMS / 16 + 7) / 8, BATCH / 64), 256, 0, stream>>>(pu16, q16, out);
}
